// PointNetSetAbstractionMsg_40785009443184
// MI455X (gfx1250) — compile-verified, matrix operand order checked by reading, not run on hardware
//
#include <hip/hip_runtime.h>

// ---------------- types ----------------
typedef _Float16 h16;
typedef __attribute__((ext_vector_type(16))) _Float16 v16h;
typedef __attribute__((ext_vector_type(8)))  _Float16 v8h;
typedef __attribute__((ext_vector_type(8)))  float    v8f;

#define B_    8
#define N_    8192
#define S_    1024
#define INCH  32
#define BNEPS 1e-5f
#define NSLOT 64   // atomic-contention spreading slots for BN stats

// ---------------- FPS: one block per batch, 1024 thr, 8 pts/thr in regs ----
__global__ __launch_bounds__(1024) void fps_kernel(const float* __restrict__ xyz,
                                                   int* __restrict__ fps_idx,
                                                   float* __restrict__ newxyz) {
  __shared__ float wval[32];
  __shared__ int   widx[32];
  __shared__ float cs[3];
  __shared__ int   sfar;
  const int b = blockIdx.x;
  const int tid = threadIdx.x;
  const int lane = tid & 31, wid = tid >> 5;
  const float* xb = xyz + (size_t)b * N_ * 3;

  float px[8], py[8], pz[8], dd[8];
#pragma unroll
  for (int j = 0; j < 8; ++j) {
    int p = tid + j * 1024;
    px[j] = xb[p * 3 + 0];
    py[j] = xb[p * 3 + 1];
    pz[j] = xb[p * 3 + 2];
    dd[j] = 1e10f;
  }

  int far = 0;
  for (int s = 0; s < S_; ++s) {
    if ((far & 1023) == tid) {
      int j = far >> 10;
      cs[0] = px[j]; cs[1] = py[j]; cs[2] = pz[j];
    }
    __syncthreads();
    float cx = cs[0], cy = cs[1], cz = cs[2];
    if (tid == 0) {
      fps_idx[b * S_ + s] = far;
      newxyz[(size_t)(b * S_ + s) * 3 + 0] = cx;
      newxyz[(size_t)(b * S_ + s) * 3 + 1] = cy;
      newxyz[(size_t)(b * S_ + s) * 3 + 2] = cz;
    }
    float bv = -1.0f; int bi = 0;
#pragma unroll
    for (int j = 0; j < 8; ++j) {
      float dx = px[j] - cx, dy = py[j] - cy, dz = pz[j] - cz;
      float d = dx * dx + dy * dy + dz * dz;
      dd[j] = fminf(dd[j], d);
      if (dd[j] > bv) { bv = dd[j]; bi = tid + j * 1024; }
    }
#pragma unroll
    for (int m = 16; m >= 1; m >>= 1) {
      float ov = __shfl_xor(bv, m);
      int   oi = __shfl_xor(bi, m);
      if (ov > bv || (ov == bv && oi < bi)) { bv = ov; bi = oi; }
    }
    if (lane == 0) { wval[wid] = bv; widx[wid] = bi; }
    __syncthreads();
    if (wid == 0) {
      bv = wval[lane]; bi = widx[lane];
#pragma unroll
      for (int m = 16; m >= 1; m >>= 1) {
        float ov = __shfl_xor(bv, m);
        int   oi = __shfl_xor(bi, m);
        if (ov > bv || (ov == bv && oi < bi)) { bv = ov; bi = oi; }
      }
      if (lane == 0) sfar = bi;
    }
    __syncthreads();
    far = sfar;
  }
}

// ---------------- ball query: one wave32 per query point ----------------
__global__ void ballquery_kernel(const float* __restrict__ xyz,
                                 const float* __restrict__ newxyz,
                                 int* __restrict__ idxbuf, float r2, int K) {
  const int lane = threadIdx.x & 31;
  const int wid  = threadIdx.x >> 5;
  const int q = blockIdx.x * (blockDim.x >> 5) + wid;  // b*S + s
  if (q >= B_ * S_) return;
  const int b = q / S_;
  const float cx = newxyz[(size_t)q * 3 + 0];
  const float cy = newxyz[(size_t)q * 3 + 1];
  const float cz = newxyz[(size_t)q * 3 + 2];
  const float* xb = xyz + (size_t)b * N_ * 3;
  int* out = idxbuf + (size_t)q * K;

  int base = 0, idx0 = 0;
  for (int n0 = 0; n0 < N_; n0 += 32) {
    int p = n0 + lane;
    float dx = xb[p * 3 + 0] - cx;
    float dy = xb[p * 3 + 1] - cy;
    float dz = xb[p * 3 + 2] - cz;
    bool pred = (dx * dx + dy * dy + dz * dz) < r2;
    unsigned mask = (unsigned)__ballot(pred);
    if (base == 0 && mask != 0u) {
      int fl = __ffs(mask) - 1;
      idx0 = __shfl(p, fl);
    }
    if (pred) {
      int pos = base + __popc(mask & ((1u << lane) - 1u));
      if (pos < K) out[pos] = p;
    }
    base += __popc(mask);
    if (base >= K) break;
  }
  for (int t = base + lane; t < K; t += 32) out[t] = idx0;
}

// -------- gather: f16 rows [M][64], float4 loads + v8h stores -------------
__global__ void gather_kernel(const float* __restrict__ xyz,
                              const float* __restrict__ feat,
                              const float* __restrict__ newxyz,
                              const int* __restrict__ idxbuf,
                              h16* __restrict__ X, int K, int M) {
  int m = blockIdx.x * blockDim.x + threadIdx.x;
  if (m >= M) return;
  int q = m / K;          // b*S + s
  int b = q / S_;
  int idx = idxbuf[m];
  const float4* f = (const float4*)(feat + ((size_t)b * N_ + idx) * INCH);
  h16* xr = X + (size_t)m * 64;
#pragma unroll
  for (int v = 0; v < 4; ++v) {
    float4 lo = f[2 * v], hi = f[2 * v + 1];
    v8h t;
    t[0] = (h16)lo.x; t[1] = (h16)lo.y; t[2] = (h16)lo.z; t[3] = (h16)lo.w;
    t[4] = (h16)hi.x; t[5] = (h16)hi.y; t[6] = (h16)hi.z; t[7] = (h16)hi.w;
    *reinterpret_cast<v8h*>(xr + v * 8) = t;
  }
  const float* p = xyz + ((size_t)b * N_ + idx) * 3;
  v8h t4 = (v8h){};
  t4[0] = (h16)(p[0] - newxyz[(size_t)q * 3 + 0]);
  t4[1] = (h16)(p[1] - newxyz[(size_t)q * 3 + 1]);
  t4[2] = (h16)(p[2] - newxyz[(size_t)q * 3 + 2]);
  *reinterpret_cast<v8h*>(xr + 32) = t4;
  v8h z = (v8h){};
  *reinterpret_cast<v8h*>(xr + 40) = z;
  *reinterpret_cast<v8h*>(xr + 48) = z;
  *reinterpret_cast<v8h*>(xr + 56) = z;
}

// ------- layer prep: f32 W -> padded f16, zero slotted BN stats -----------
__global__ void prep_layer_kernel(const float* __restrict__ W, h16* __restrict__ Wh,
                                  float* __restrict__ stats, int O, int Cin, int Cp) {
  int i = blockIdx.x * blockDim.x + threadIdx.x;
  if (i < NSLOT * 2 * O) stats[i] = 0.0f;
  if (i < O * Cp) {
    int o = i / Cp, c = i % Cp;
    Wh[i] = (c < Cin) ? (h16)W[o * Cin + c] : (h16)0.0f;
  }
}

// ---- WMMA GEMM: 8 waves/block, 1 wave = 16-pt tile x ALL O tiles ---------
// B fragment loaded once per K-chunk, reused across OT WMMAs. If BN=true the
// previous layer's BN+ReLU (per-channel affine + max0) is applied while
// unpacking the B fragment (exact fusion: each X row is read once per wave).
template <int KT, int OT, bool BN>
__global__ __launch_bounds__(256) void gemm_wmma_kernel(
    const h16* __restrict__ Wh, const float* __restrict__ bias,
    const float* __restrict__ ss,  // [Cp] scale, [Cp] shift (input-side BN)
    const h16* __restrict__ X, h16* __restrict__ Y) {
  const int lane = threadIdx.x & 31;
  const int w    = threadIdx.x >> 5;
  const int n = lane & 15;   // D col / B col / A row within tile
  const int g = lane >> 4;   // K-half selector
  const int p = (blockIdx.x * 8 + w) * 16 + n;   // point index
  const int Cp = KT * 32;
  const int O  = OT * 16;

  const h16* brow = X + (size_t)p * Cp;
  v8f acc[OT];
#pragma unroll
  for (int t = 0; t < OT; ++t) acc[t] = (v8f){};

#pragma unroll
  for (int kc = 0; kc < KT; ++kc) {
    const h16* pb = brow + kc * 32 + g * 8;
    v8h b0 = *reinterpret_cast<const v8h*>(pb);
    v8h b1 = *reinterpret_cast<const v8h*>(pb + 16);
    if (BN) {  // apply previous layer's BN + ReLU on the fly
      const int c0 = kc * 32 + g * 8;
      const int c1 = c0 + 16;
#pragma unroll
      for (int i = 0; i < 8; ++i) {
        float x0 = (float)b0[i];
        float x1 = (float)b1[i];
        x0 = fmaxf(x0 * ss[c0 + i] + ss[Cp + c0 + i], 0.0f);
        x1 = fmaxf(x1 * ss[c1 + i] + ss[Cp + c1 + i], 0.0f);
        b0[i] = (h16)x0;
        b1[i] = (h16)x1;
      }
    }
    v16h bm;
#pragma unroll
    for (int i = 0; i < 8; ++i) { bm[i] = b0[i]; bm[i + 8] = b1[i]; }
#pragma unroll
    for (int t = 0; t < OT; ++t) {
      const h16* pa = Wh + (size_t)(t * 16 + n) * Cp + kc * 32 + g * 8;
      v8h a0 = *reinterpret_cast<const v8h*>(pa);
      v8h a1 = *reinterpret_cast<const v8h*>(pa + 16);
      v16h a;
#pragma unroll
      for (int i = 0; i < 8; ++i) { a[i] = a0[i]; a[i + 8] = a1[i]; }
      acc[t] = __builtin_amdgcn_wmma_f32_16x16x32_f16(
          /*neg_a=*/false, a, /*neg_b=*/false, bm,
          /*c_mod=*/(short)0, acc[t], /*reuse_a=*/false, /*reuse_b=*/false);
    }
  }

  // lean epilogue: bias + f16 store only (BN stats done by stats_kernel)
#pragma unroll
  for (int t = 0; t < OT; ++t) {
    const int obase = t * 16 + g * 8;
    v8h outv;
#pragma unroll
    for (int r = 0; r < 8; ++r) outv[r] = (h16)(acc[t][r] + bias[obase + r]);
    *reinterpret_cast<v8h*>(Y + (size_t)p * O + obase) = outv;
  }
}

// ---- BN statistics: streaming v8h reads, LDS reduce, slotted atomics -----
__global__ __launch_bounds__(256) void stats_kernel(const h16* __restrict__ Y,
                                                    float* __restrict__ stats,
                                                    long M, int O, int G /*O/8*/) {
  __shared__ float acc[2 * 128];
  const int tid = threadIdx.x;
  for (int i = tid; i < 2 * O; i += 256) acc[i] = 0.0f;
  __syncthreads();

  const int cg = tid % G;        // column group (8 channels)
  const int rl = tid / G;        // row lane within block
  const int rows = 256 / G;      // rows handled per block per stride
  if (rl < rows) {
    float s1[8] = {0, 0, 0, 0, 0, 0, 0, 0};
    float s2[8] = {0, 0, 0, 0, 0, 0, 0, 0};
    for (long m = (long)blockIdx.x * rows + rl; m < M; m += (long)rows * gridDim.x) {
      v8h v = *reinterpret_cast<const v8h*>(Y + m * O + cg * 8);
#pragma unroll
      for (int r = 0; r < 8; ++r) {
        float x = (float)v[r];
        s1[r] += x;
        s2[r] += x * x;
      }
    }
#pragma unroll
    for (int r = 0; r < 8; ++r) {
      atomicAdd(&acc[cg * 8 + r], s1[r]);
      atomicAdd(&acc[O + cg * 8 + r], s2[r]);
    }
  }
  __syncthreads();
  float* slot = stats + (size_t)(blockIdx.x & (NSLOT - 1)) * (2 * O);
  for (int i = tid; i < 2 * O; i += 256) atomicAdd(&slot[i], acc[i]);
}

// ------- BN finalize: sum slots, produce per-channel scale/shift ----------
__global__ void bn_finalize_kernel(const float* __restrict__ stats,
                                   const float* __restrict__ gamma,
                                   const float* __restrict__ beta,
                                   float* __restrict__ ss, int O, float invM) {
  int o = blockIdx.x * blockDim.x + threadIdx.x;
  if (o < O) {
    float s = 0.0f, q = 0.0f;
    for (int j = 0; j < NSLOT; ++j) {
      s += stats[(size_t)j * 2 * O + o];
      q += stats[(size_t)j * 2 * O + O + o];
    }
    float mean = s * invM;
    float var  = q * invM - mean * mean;
    float sc   = gamma[o] * rsqrtf(var + BNEPS);
    ss[o]     = sc;
    ss[O + o] = beta[o] - mean * sc;
  }
}

// ------- fused BN + ReLU + max-pool over K, write f32 into d_out feat -----
__global__ void pool_bn_kernel(const h16* __restrict__ Y, const float* __restrict__ ss,
                               float* __restrict__ out_feat,
                               int K, int O, int choff, int total) {
  int t = blockIdx.x * blockDim.x + threadIdx.x;
  if (t >= total) return;
  int o = t % O;
  int q = t / O;  // b*S + s
  const float sc = ss[o];
  const float sh = ss[O + o];
  const h16* p = Y + ((size_t)q * K) * O + o;
  float m = -3.4e38f;
  for (int k = 0; k < K; ++k) {
    if ((k & 15) == 0 && k + 16 < K) __builtin_prefetch(p + (size_t)(k + 16) * O, 0, 0);
    float x = (float)p[(size_t)k * O];
    m = fmaxf(m, x * sc + sh);
  }
  out_feat[(size_t)q * 320 + choff + o] = fmaxf(m, 0.0f);  // ReLU commutes w/ max
}

// ==========================================================================
static void launch_gemm(int KT, int OT, bool bn, int M,
                        const h16* Wh, const float* bias, const float* ss,
                        const h16* Xin, h16* Xout, hipStream_t stream) {
  dim3 grid(M / 128);  // 8 waves/block, 16 pts/wave
#define GEMM_CASE(kt, ot)                                                      \
  case ((kt) * 16 + (ot)):                                                     \
    if (bn)                                                                    \
      gemm_wmma_kernel<kt, ot, true><<<grid, 256, 0, stream>>>(Wh, bias, ss,   \
                                                               Xin, Xout);     \
    else                                                                       \
      gemm_wmma_kernel<kt, ot, false><<<grid, 256, 0, stream>>>(Wh, bias, ss,  \
                                                                Xin, Xout);    \
    break;
  switch (KT * 16 + OT) {
    GEMM_CASE(1, 2) GEMM_CASE(1, 4) GEMM_CASE(1, 6) GEMM_CASE(1, 8)
    GEMM_CASE(2, 2) GEMM_CASE(2, 4) GEMM_CASE(2, 6) GEMM_CASE(2, 8)
    GEMM_CASE(3, 2) GEMM_CASE(3, 4) GEMM_CASE(3, 6) GEMM_CASE(3, 8)
    GEMM_CASE(4, 2) GEMM_CASE(4, 4) GEMM_CASE(4, 6) GEMM_CASE(4, 8)
    default: break;
  }
#undef GEMM_CASE
}

extern "C" void kernel_launch(void* const* d_in, const int* in_sizes, int n_in,
                              void* d_out, int out_size, void* d_ws, size_t ws_size,
                              hipStream_t stream) {
  const float* xyz      = (const float*)d_in[0];
  const float* features = (const float*)d_in[1];
  // params flattened depth-first: branch -> layer -> (W, b, gamma, beta)
  auto Wp = [&](int br, int ly) { return (const float*)d_in[2 + (br * 3 + ly) * 4 + 0]; };
  auto bp = [&](int br, int ly) { return (const float*)d_in[2 + (br * 3 + ly) * 4 + 1]; };
  auto gp = [&](int br, int ly) { return (const float*)d_in[2 + (br * 3 + ly) * 4 + 2]; };
  auto zp = [&](int br, int ly) { return (const float*)d_in[2 + (br * 3 + ly) * 4 + 3]; };

  float* out_newxyz = (float*)d_out;                        // (B,S,3)
  float* out_feat   = (float*)d_out + (size_t)B_ * S_ * 3;  // (B,S,320)

  // workspace carve (256B aligned)
  char* ws = (char*)d_ws;
  size_t off = 0;
  auto carve = [&](size_t bytes) {
    void* p = ws + off;
    off = (off + bytes + 255) & ~(size_t)255;
    return p;
  };
  int*   fps_idx = (int*)carve((size_t)B_ * S_ * sizeof(int));
  int*   idxbuf  = (int*)carve((size_t)B_ * S_ * 128 * sizeof(int));
  h16*   Wh      = (h16*)carve((size_t)128 * 128 * sizeof(h16));
  float* stats   = (float*)carve((size_t)NSLOT * 2 * 128 * sizeof(float));
  float* ssbuf   = (float*)carve((size_t)2 * 128 * sizeof(float));
  const size_t MMAX = (size_t)B_ * S_ * 128;  // largest branch point count
  h16* bufs[2];
  bufs[0] = (h16*)carve(MMAX * 128 * sizeof(h16));
  bufs[1] = (h16*)carve(MMAX * 128 * sizeof(h16));

  // 1) FPS (writes fps_idx + new_xyz straight into d_out)
  fps_kernel<<<B_, 1024, 0, stream>>>(xyz, fps_idx, out_newxyz);

  const int   KS[3]      = {16, 32, 128};
  const float R2[3]      = {0.01f, 0.04f, 0.16f};
  const int   MLPS[3][3] = {{32, 32, 64}, {64, 64, 128}, {64, 96, 128}};
  const int   CHOFF[3]   = {0, 64, 192};

  for (int br = 0; br < 3; ++br) {
    const int K = KS[br];
    const int M = B_ * S_ * K;

    // 2) ball query: one wave per query, 8 waves/block
    ballquery_kernel<<<(B_ * S_) / 8, 256, 0, stream>>>(xyz, out_newxyz, idxbuf, R2[br], K);

    // 3) gather into bufs[0] as f16 [M][64]
    gather_kernel<<<(M + 255) / 256, 256, 0, stream>>>(xyz, features, out_newxyz,
                                                       idxbuf, bufs[0], K, M);

    // 4) MLP layers (ping-pong bufs[0]/bufs[1]); input-side BN fused into GEMM
    for (int ly = 0; ly < 3; ++ly) {
      const int Cin = (ly == 0) ? 35 : MLPS[br][ly - 1];
      const int Cp  = (ly == 0) ? 64 : MLPS[br][ly - 1];
      const int O   = MLPS[br][ly];
      h16* Xin  = bufs[ly & 1];
      h16* Xout = bufs[(ly & 1) ^ 1];

      int prepElems = (O * Cp > NSLOT * 2 * O) ? O * Cp : NSLOT * 2 * O;
      prep_layer_kernel<<<(prepElems + 255) / 256, 256, 0, stream>>>(
          Wp(br, ly), Wh, stats, O, Cin, Cp);

      // GEMM consumes ssbuf from the PREVIOUS layer's finalize (ly>0 only)
      launch_gemm(Cp / 32, O / 16, /*bn=*/(ly > 0), M, Wh, bp(br, ly), ssbuf,
                  Xin, Xout, stream);

      // global BN stats on pre-BN output, then scale/shift for next consumer
      stats_kernel<<<1024, 256, 0, stream>>>(Xout, stats, (long)M, O, O / 8);
      bn_finalize_kernel<<<1, 128, 0, stream>>>(stats, gp(br, ly), zp(br, ly),
                                                ssbuf, O, 1.0f / (float)M);
    }

    // 5) fused BN + ReLU + max-pool over K -> d_out feature block
    const int Olast = MLPS[br][2];
    int total = B_ * S_ * Olast;
    pool_bn_kernel<<<(total + 255) / 256, 256, 0, stream>>>(bufs[1], ssbuf, out_feat,
                                                            K, Olast, CHOFF[br], total);
  }
  (void)in_sizes; (void)n_in; (void)out_size; (void)ws_size;
}
